// Transformer_3315714752562
// MI455X (gfx1250) — matrix operand order repaired and checked by reading, not compiled, not run on hardware
//
#include <hip/hip_runtime.h>
#include <math.h>

#define N_TOK 16384
#define D_IN  256
#define M_DIM 64
#define TW    32          // tokens per workgroup
#define LSTR  72          // LDS f32 row stride (floats)
#define HSTR  72          // LDS f16 shadow row stride (halves)

typedef __attribute__((ext_vector_type(16))) _Float16 v16h;
typedef __attribute__((ext_vector_type(8)))  _Float16 v8h;
typedef __attribute__((ext_vector_type(8)))  float    v8f;
typedef unsigned int v4u __attribute__((ext_vector_type(4)));
typedef int          v8i __attribute__((ext_vector_type(8)));
typedef int          v4i __attribute__((ext_vector_type(4)));

__device__ __forceinline__ float leakyf(float v, float s) {
    return v >= 0.0f ? v : s * v;
}

// One 16x16 output tile per wave. A from f16 shadow in LDS (row-major, stride HSTR),
// B from LDS pre-packed WMMA-B fragment layout (f16). D epilogue: +bias, optional
// leaky(0.01), optional residual add, stores f32 master + optional f16 shadow.
__device__ __forceinline__ void gemm64_tile(const _Float16* __restrict__ Ash,
                                            const _Float16* __restrict__ Wpk,
                                            const float* __restrict__ bias,
                                            float* __restrict__ Dst,
                                            _Float16* __restrict__ Dsth,
                                            const float* __restrict__ addSrc,
                                            int wave, int lane, int leakyMode)
{
    const int rt  = wave >> 2;
    const int ct  = wave & 3;
    const int row = lane & 15;
    const int hi  = lane >> 4;
    const int col = ct * 16 + row;

    v8f acc = {};
#pragma unroll
    for (int kt = 0; kt < 2; ++kt) {
        const _Float16* ap = Ash + (rt * 16 + row) * HSTR + kt * 32 + hi * 16;
        v8h a0 = *(const v8h*)ap;           // K = kb .. kb+7
        v8h a1 = *(const v8h*)(ap + 8);     // K = kb+8 .. kb+15
        v16h a = __builtin_shufflevector(a0, a1, 0, 1, 2, 3, 4, 5, 6, 7,
                                                 8, 9, 10, 11, 12, 13, 14, 15);
        v16h b = *(const v16h*)(Wpk + ((((ct << 1) | kt) * 32) + lane) * 16);
        acc = __builtin_amdgcn_wmma_f32_16x16x32_f16(false, a, false, b,
                                                     (short)0, acc, false, false);
    }

    const float bvv = bias[col];
#pragma unroll
    for (int i = 0; i < 8; ++i) {
        const int r = rt * 16 + hi * 8 + i;
        float v = acc[i] + bvv;
        if (leakyMode) v = leakyf(v, 0.01f);
        if (addSrc) v += addSrc[r * LSTR + col];
        Dst[r * LSTR + col] = v;
        if (Dsth) Dsth[r * HSTR + col] = (_Float16)v;
    }
}

__global__ __launch_bounds__(256)
void fused_transformer_kernel(const float* __restrict__ x,
                              const float* __restrict__ W1,  const float* __restrict__ b1,
                              const float* __restrict__ Wq,  const float* __restrict__ bq,
                              const float* __restrict__ Wk,  const float* __restrict__ bk,
                              const float* __restrict__ Wv,  const float* __restrict__ bv,
                              const float* __restrict__ Wf1, const float* __restrict__ bf1,
                              const float* __restrict__ Wf2, const float* __restrict__ bf2,
                              const float* __restrict__ ln_g, const float* __restrict__ ln_b,
                              const float* __restrict__ Wout, const float* __restrict__ bout,
                              float* __restrict__ out)
{
    __shared__ float    s_x[TW * D_IN];        // TDM-staged x tile (32x256 fp32)
    __shared__ float    s_cur[TW * LSTR];      // block input (f32 master)
    __shared__ float    s_res[TW * LSTR];      // running residual
    __shared__ float    s_A[TW * LSTR];        // q, later f1 hidden
    __shared__ float    s_B[TW * LSTR];        // k, later pre-LN output
    __shared__ float    s_C[TW * LSTR];        // v
    __shared__ float    s_D[TW * LSTR];        // y
    __shared__ _Float16 s_curh[TW * HSTR];     // f16 shadows of GEMM A-sources
    __shared__ _Float16 s_Ah[TW * HSTR];
    __shared__ _Float16 s_Dh[TW * HSTR];
    __shared__ _Float16 s_wpk[5 * M_DIM * M_DIM];  // WMMA-B packed weights (one block)
    __shared__ float    s_kmx[TW];
    __shared__ float    s_kmn[TW];

    const int tid  = threadIdx.x;
    const int wave = tid >> 5;
    const int lane = tid & 31;
    const int n0   = blockIdx.x * TW;

    // ------------- stage x tile into LDS (Tensor Data Mover, 2D 32x256 fp32) -------------
#if __has_builtin(__builtin_amdgcn_tensor_load_to_lds)
    {
        if (wave == 0) {
            const unsigned long long gaddr =
                (unsigned long long)(const void*)(x + (size_t)n0 * D_IN);
            const unsigned int lds_off = (unsigned int)(unsigned long long)(void*)&s_x[0];
            v4u g0;
            g0[0] = 1u;                                   // count=1, user D#
            g0[1] = lds_off;                              // lds_addr
            g0[2] = (unsigned int)gaddr;                  // global_addr[31:0]
            g0[3] = (unsigned int)((gaddr >> 32) & 0x1FFFFFFull) | (2u << 30); // addr[56:32], type=2
            v8i g1;
            g1[0] = (int)(2u << 16);                      // wg_mask=0, data_size=4B
            g1[1] = (int)(((unsigned)D_IN & 0xFFFFu) << 16);          // tensor_dim0 lo16
            g1[2] = (int)(((unsigned)N_TOK & 0xFFFFu) << 16);         // td0 hi16 | tensor_dim1 lo16
            g1[3] = (int)((unsigned)D_IN << 16);          // td1 hi16 | tile_dim0 = 256
            g1[4] = (int)TW;                              // tile_dim1 = 32, tile_dim2 = 0
            g1[5] = (int)D_IN;                            // tensor_dim0_stride lo32 = 256
            g1[6] = 0;                                    // stride0 hi16 | stride1 lo16
            g1[7] = 0;                                    // stride1 hi32
            v4i g2 = {0, 0, 0, 0};
            v4i g3 = {0, 0, 0, 0};
#if defined(__clang_major__) && (__clang_major__ >= 23)
            v8i gx = {0, 0, 0, 0, 0, 0, 0, 0};
            __builtin_amdgcn_tensor_load_to_lds(g0, g1, g2, g3, gx, 0);
#else
            __builtin_amdgcn_tensor_load_to_lds(g0, g1, g2, g3, 0);
#endif
            __builtin_amdgcn_s_wait_tensorcnt(0);
        }
    }
#else
    {
#pragma unroll
        for (int i = 0; i < (TW * D_IN) / 256; ++i)
            s_x[tid + i * 256] = x[(size_t)n0 * D_IN + tid + i * 256];
    }
#endif
    __syncthreads();

    // ---------------- Layer 1: h = x @ W1 + b1, K = 256 ----------------
    {
        const int rt  = wave >> 2;
        const int ct  = wave & 3;
        const int row = lane & 15;
        const int hi  = lane >> 4;
        const int col = ct * 16 + row;
        const float* xrow = s_x + (rt * 16 + row) * D_IN;

        v8f acc = {};
#pragma unroll
        for (int kt = 0; kt < 8; ++kt) {
            const int kb = kt * 32 + hi * 16;
            v16h a;
#pragma unroll
            for (int t = 0; t < 8; ++t) a[t]     = (_Float16)xrow[kb + t];
#pragma unroll
            for (int t = 0; t < 8; ++t) a[8 + t] = (_Float16)xrow[kb + 8 + t];
            const int kw = kt * 32 + hi * 16;
            v16h b;
#pragma unroll
            for (int t = 0; t < 16; ++t)
                b[t] = (_Float16)W1[(kw + t) * M_DIM + col];
            acc = __builtin_amdgcn_wmma_f32_16x16x32_f16(false, a, false, b,
                                                         (short)0, acc, false, false);
        }
        const float bvv = b1[col];
#pragma unroll
        for (int i = 0; i < 8; ++i) {
            const int r = rt * 16 + hi * 8 + i;
            const float h = acc[i] + bvv;
            s_cur[r * LSTR + col]  = h;                    // transformer input (pre-act)
            s_curh[r * HSTR + col] = (_Float16)h;
            s_res[r * LSTR + col]  = leakyf(h, 0.1f);      // x1 = LeakyReLU(0.1)(h)
        }
    }
    __syncthreads();

    // ---------------- 3 stacked attention blocks ----------------
    for (int t = 0; t < 3; ++t) {
        // ---- pack this block's 5 weight matrices into WMMA-B f16 fragment layout ----
        {
            const float* Wsrc[5] = { Wq  + t * M_DIM * M_DIM, Wk  + t * M_DIM * M_DIM,
                                     Wv  + t * M_DIM * M_DIM, Wf1 + t * M_DIM * M_DIM,
                                     Wf2 + t * M_DIM * M_DIM };
#pragma unroll
            for (int m = 0; m < 5; ++m) {
                const float* W = Wsrc[m];
#pragma unroll
                for (int i = 0; i < 16; ++i) {
                    const int e   = tid + i * 256;     // 0..4095, coalesced global read
                    const int k   = e >> 6;
                    const int col = e & 63;
                    const int ct2 = col >> 4, c = col & 15;
                    const int kt2 = k >> 5, hi2 = (k >> 4) & 1, tt = k & 15;
                    const int ln  = hi2 * 16 + c;
                    s_wpk[m * M_DIM * M_DIM +
                          (((ct2 << 1) | kt2) * 32 + ln) * 16 + tt] = (_Float16)W[e];
                }
            }
        }
        __syncthreads();

        // q, k, v projections
        gemm64_tile(s_curh, s_wpk + 0 * M_DIM * M_DIM, bq + t * M_DIM, s_A, nullptr, nullptr, wave, lane, 0);
        gemm64_tile(s_curh, s_wpk + 1 * M_DIM * M_DIM, bk + t * M_DIM, s_B, nullptr, nullptr, wave, lane, 0);
        gemm64_tile(s_curh, s_wpk + 2 * M_DIM * M_DIM, bv + t * M_DIM, s_C, nullptr, nullptr, wave, lane, 0);
        __syncthreads();

        // per-token k max/min (max_j q_i*k_j = q_i * (q_i>=0 ? kmax : kmin))
        if (tid < TW) {
            float mx = -INFINITY, mn = INFINITY;
#pragma unroll 8
            for (int j = 0; j < M_DIM; ++j) {
                const float kv = s_B[tid * LSTR + j];
                mx = fmaxf(mx, kv);
                mn = fminf(mn, kv);
            }
            s_kmx[tid] = mx;
            s_kmn[tid] = mn;
        }
        __syncthreads();

        // attention: y[n,i] = softmax_j(q_i*k_j) . v  + cur[n,i]
#pragma unroll
        for (int p = 0; p < 8; ++p) {
            const int r = tid + p * 256;
            const int n = r >> 6;
            const int i = r & 63;
            const float qi = s_A[n * LSTR + i];
            const float m  = (qi >= 0.0f) ? qi * s_kmx[n] : qi * s_kmn[n];
            float s = 0.0f, yv = 0.0f;
#pragma unroll 8
            for (int j = 0; j < M_DIM; ++j) {
                const float e = __expf(qi * s_B[n * LSTR + j] - m);
                s  += e;
                yv += e * s_C[n * LSTR + j];
            }
            const float yval = yv / s + s_cur[n * LSTR + i];
            s_D[n * LSTR + i]  = yval;
            s_Dh[n * HSTR + i] = (_Float16)yval;
        }
        __syncthreads();

        // h1 = LeakyReLU(0.01)(y @ Wf1 + bf1)   (shadow needed: it is f2's A-source)
        gemm64_tile(s_Dh, s_wpk + 3 * M_DIM * M_DIM, bf1 + t * M_DIM, s_A, s_Ah, nullptr, wave, lane, 1);
        __syncthreads();
        // a = h1 @ Wf2 + bf2 + y
        gemm64_tile(s_Ah, s_wpk + 4 * M_DIM * M_DIM, bf2 + t * M_DIM, s_B, nullptr, s_D, wave, lane, 0);
        __syncthreads();

        // LayerNorm(a) + res -> res ; cur = res (f32 + f16 shadow)
        if (tid < TW) {
            const float* arow = s_B + tid * LSTR;
            float mu = 0.0f;
#pragma unroll 8
            for (int j = 0; j < M_DIM; ++j) mu += arow[j];
            mu *= (1.0f / M_DIM);
            float var = 0.0f;
#pragma unroll 8
            for (int j = 0; j < M_DIM; ++j) {
                const float d = arow[j] - mu;
                var += d * d;
            }
            var *= (1.0f / M_DIM);
            const float inv = rsqrtf(var + 1e-5f);
#pragma unroll 8
            for (int j = 0; j < M_DIM; ++j) {
                const float lnv = (arow[j] - mu) * inv * ln_g[t * M_DIM + j] + ln_b[t * M_DIM + j];
                const float nr  = lnv + s_res[tid * LSTR + j];
                s_res[tid * LSTR + j]  = nr;
                s_cur[tid * LSTR + j]  = nr;
                s_curh[tid * HSTR + j] = (_Float16)nr;
            }
        }
        __syncthreads();
    }

    // ---------------- output head: leaky(res @ Wout + bout, 0.1), 5 copies ----------------
    if (tid < TW) {
        float acc = bout[0];
#pragma unroll 8
        for (int j = 0; j < M_DIM; ++j) acc += s_res[tid * LSTR + j] * Wout[j];
        acc = leakyf(acc, 0.1f);
        const int n = n0 + tid;
#pragma unroll
        for (int c = 0; c < 5; ++c) out[c * N_TOK + n] = acc;
    }
}

extern "C" void kernel_launch(void* const* d_in, const int* in_sizes, int n_in,
                              void* d_out, int out_size, void* d_ws, size_t ws_size,
                              hipStream_t stream) {
    (void)in_sizes; (void)n_in; (void)out_size; (void)d_ws; (void)ws_size;
    const float* x    = (const float*)d_in[0];
    const float* W1   = (const float*)d_in[1];
    const float* b1   = (const float*)d_in[2];
    const float* Wq   = (const float*)d_in[3];
    const float* bq   = (const float*)d_in[4];
    const float* Wk   = (const float*)d_in[5];
    const float* bk   = (const float*)d_in[6];
    const float* Wv   = (const float*)d_in[7];
    const float* bv   = (const float*)d_in[8];
    const float* Wf1  = (const float*)d_in[9];
    const float* bf1  = (const float*)d_in[10];
    const float* Wf2  = (const float*)d_in[11];
    const float* bf2  = (const float*)d_in[12];
    const float* ln_g = (const float*)d_in[13];
    const float* ln_b = (const float*)d_in[14];
    const float* Wout = (const float*)d_in[15];
    const float* bout = (const float*)d_in[16];
    float* out = (float*)d_out;

    dim3 grid(N_TOK / TW);
    dim3 block(256);
    fused_transformer_kernel<<<grid, block, 0, stream>>>(
        x, W1, b1, Wq, bq, Wk, bk, Wv, bv, Wf1, bf1, Wf2, bf2,
        ln_g, ln_b, Wout, bout, out);
}
